// RNNTJoiner_17334488007387
// MI455X (gfx1250) — compile-verified, matrix operand order checked by reading, not run on hardware
//
#include <hip/hip_runtime.h>
#include <hip/hip_bf16.h>
#include <math.h>

// ---------------------------------------------------------------------------
// RNN-T joiner for MI455X (gfx1250):
//   e = enc @ W_enc + b_enc              (1600 x 512)   fp32, tiny
//   p = pred @ W_pred + b_pred           ( 800 x 512)   fp32, tiny
//   logits = tanh(e (+) p) @ W_out + b   (160000 x 512) @ (512 x 1024)
// Big GEMM: v_wmma_f32_16x16x32_f16, 128x1024 C-patch per block, 4x4 WMMA
// outer product per wave per K-step (16 WMMAs : 4 LDS + 4 global loads).
// Tail (u) handling is hoisted to a block-uniform branch so 6/7 of blocks
// run fully unpredicated fill/store paths.
// ---------------------------------------------------------------------------

typedef _Float16 v16h __attribute__((ext_vector_type(16)));
typedef _Float16 v8h  __attribute__((ext_vector_type(8)));
typedef float    v8f  __attribute__((ext_vector_type(8)));

#define RB   8        // batch
#define RT   200      // T
#define RU   100      // U
#define RE   512      // ENC_DIM
#define RP   640      // PRED_DIM
#define RJ   512      // JOINT_DIM
#define RV   1024     // VOCAB
#define UT   7        // ceil(U/16)
#define TG   25       // T / 8 (t-groups of 8 per block)
#define ASTR 520      // padded LDS row stride in halfs (1040 B: breaks bank conflicts)

__device__ __forceinline__ float tanh_fast(float x) {
    // tanh(x) = 1 - 2/(exp(2x)+1); exact at +/-inf, ~1e-3 accurate (f16 operand)
    const float e2 = __expf(2.0f * x);
    return 1.0f - 2.0f * __builtin_amdgcn_rcpf(e2 + 1.0f);
}

// ---------------------------------------------------------------------------
// Stage 1: thin projections e = enc*W_enc+b_enc, p = pred*W_pred+b_pred (fp32)
// ---------------------------------------------------------------------------
__global__ __launch_bounds__(256) void rnnt_proj(
    const float* __restrict__ enc,  const float* __restrict__ pred,
    const float* __restrict__ W_enc, const float* __restrict__ b_enc,
    const float* __restrict__ W_pred, const float* __restrict__ b_pred,
    float* __restrict__ e_ws, float* __restrict__ p_ws)
{
    __shared__ float x_s[RP];   // max(RE, RP) = 640
    const int tid = threadIdx.x;

    const float* x;  const float* W;  const float* bias;  float* dst;  int K;
    if (blockIdx.x < RB * RT) {
        const int row = blockIdx.x;
        x = enc + (size_t)row * RE;  W = W_enc;  bias = b_enc;
        dst = e_ws + (size_t)row * RJ;  K = RE;
    } else {
        const int row = blockIdx.x - RB * RT;
        x = pred + (size_t)row * RP;  W = W_pred;  bias = b_pred;
        dst = p_ws + (size_t)row * RJ;  K = RP;
    }

    for (int i = tid; i < K; i += 256) x_s[i] = x[i];
    __syncthreads();

    float a0 = 0.f, a1 = 0.f;
    for (int h = 0; h < K; ++h) {
        const float xv = x_s[h];
        a0 = fmaf(xv, W[(size_t)h * RJ + tid],       a0);
        a1 = fmaf(xv, W[(size_t)h * RJ + tid + 256], a1);
    }
    dst[tid]       = a0 + bias[tid];
    dst[tid + 256] = a1 + bias[tid + 256];
}

// ---------------------------------------------------------------------------
// Stage 2: W_out (512x1024 fp32, K x N) -> f16 WMMA B-fragments:
//   frag index = ((kb*64 + nt)*32 + lane), 16 halfs each
//   lane: n = nt*16 + (lane&15), half = lane>>4
//   elems 0..7  : K = kb*32 +      8*half + i
//   elems 8..15 : K = kb*32 + 16 + 8*half + i
// ---------------------------------------------------------------------------
__global__ __launch_bounds__(256) void rnnt_wswz(
    const float* __restrict__ W_out, _Float16* __restrict__ Wf)
{
    const int g    = blockIdx.x * 256 + threadIdx.x;   // 0 .. 16*64*32-1
    const int kb   = g >> 11;
    const int rem  = g & 2047;
    const int nt   = rem >> 5;
    const int lane = rem & 31;
    const int half = lane >> 4;
    const int n    = nt * 16 + (lane & 15);

    v16h v;
#pragma unroll
    for (int i = 0; i < 8; ++i) {
        const int k0 = kb * 32 +      8 * half + i;
        const int k1 = kb * 32 + 16 + 8 * half + i;
        v[i]     = (_Float16)W_out[(size_t)k0 * RV + n];
        v[i + 8] = (_Float16)W_out[(size_t)k1 * RV + n];
    }
    ((v16h*)Wf)[g] = v;
}

// ---------------------------------------------------------------------------
// Store helper: FULL=true -> unpredicated (block-uniform fast path)
// ---------------------------------------------------------------------------
template <bool FULL>
__device__ __forceinline__ void store_chunk(
    float* __restrict__ out, const float* __restrict__ b_out,
    const v8f* acc, int ch, int b, int t0, int wm, int wn,
    int u0, int m, int half)
{
#pragma unroll
    for (int i = 0; i < 4; ++i) {
        const int t = t0 + (4 * wm + i);
        const size_t rowb = ((size_t)(b * RT + t) * RU) * RV;
#pragma unroll
        for (int j = 0; j < 4; ++j) {
            const int   n    = (ch * 16 + wn * 4 + j) * 16 + m;
            const float bias = b_out[n];
#pragma unroll
            for (int r = 0; r < 8; ++r) {
                const int u = u0 + r + 8 * half;
                if (FULL || u < RU)
                    out[rowb + (size_t)u * RV + n] = acc[i * 4 + j][r] + bias;
            }
        }
    }
}

// ---------------------------------------------------------------------------
// Stage 3: fused tanh + GEMM with WMMA.
// Block: (b, t-group of 8, u-tile of 16) -> A tile 128x512 f16 in LDS (rows
// r = ti*16 + urow). 8 waves: wave (wm = w>>2, wn = w&3) computes a 64x64
// sub-patch per N-chunk (4 m-tiles x 4 n-tiles), looping 4 chunks over V=1024.
// ---------------------------------------------------------------------------
__global__ __launch_bounds__(256) void rnnt_joint(
    const float* __restrict__ e_ws, const float* __restrict__ p_ws,
    const _Float16* __restrict__ Wf, const float* __restrict__ b_out,
    float* __restrict__ out)
{
    const int tid = threadIdx.x;
    const int ut  = blockIdx.x % UT;
    int tmp       = blockIdx.x / UT;
    const int tg  = tmp % TG;
    const int b   = tmp / TG;
    const int t0  = tg * 8;
    const int u0  = ut * 16;
    const int bt0 = b * RT + t0;
    const bool fullTile = (u0 + 16 <= RU);   // uniform: false only for ut==6

    __shared__ float e_s[8 * RJ];                       // 16 KB
    __shared__ __align__(16) _Float16 A_s[128 * ASTR];  // 133 KB

    // 8 e-rows -> LDS
    for (int idx = tid; idx < 8 * RJ; idx += 256) {
        const int ti = idx >> 9, j = idx & (RJ - 1);
        e_s[idx] = e_ws[(size_t)(bt0 + ti) * RJ + j];
    }
    __syncthreads();

    // A tile: rows r = ti*16 + urow, cols j: tanh(e[ti][j] + p[u0+urow][j])
    const float* pbase = p_ws + ((size_t)b * RU + u0) * RJ;
    if (fullTile) {
        for (int idx = tid; idx < 128 * RJ; idx += 256) {
            const int r = idx >> 9, j = idx & (RJ - 1);
            A_s[r * ASTR + j] = (_Float16)tanh_fast(
                e_s[(r >> 4) * RJ + j] + pbase[(size_t)(r & 15) * RJ + j]);
        }
    } else {
        for (int idx = tid; idx < 128 * RJ; idx += 256) {
            const int r = idx >> 9, j = idx & (RJ - 1);
            float v = 0.f;
            if (u0 + (r & 15) < RU)
                v = tanh_fast(e_s[(r >> 4) * RJ + j] +
                              pbase[(size_t)(r & 15) * RJ + j]);
            A_s[r * ASTR + j] = (_Float16)v;
        }
    }
    __syncthreads();

    const int w    = tid >> 5;
    const int lane = tid & 31;
    const int wm   = w >> 2;       // 0..1 : m-group (rows 64*wm .. +64)
    const int wn   = w & 3;        // 0..3 : n-group within chunk
    const int half = lane >> 4;
    const int m    = lane & 15;

#pragma unroll 1
    for (int ch = 0; ch < 4; ++ch) {
        v8f acc[16];
        const v8f zero = {};
#pragma unroll
        for (int i = 0; i < 16; ++i) acc[i] = zero;

        for (int kk = 0; kk < 16; ++kk) {
            v16h a[4], bb[4];
#pragma unroll
            for (int i = 0; i < 4; ++i) {
                const _Float16* ap =
                    A_s + ((4 * wm + i) * 16 + m) * ASTR + kk * 32 + 8 * half;
                const v8h lo = *(const v8h*)ap;
                const v8h hi = *(const v8h*)(ap + 16);
                a[i] = __builtin_shufflevector(lo, hi,
                         0,1,2,3,4,5,6,7,8,9,10,11,12,13,14,15);
            }
#pragma unroll
            for (int j = 0; j < 4; ++j) {
                const int ntg = ch * 16 + wn * 4 + j;   // global n-tile
                bb[j] = *((const v16h*)Wf + (size_t)(kk * 64 + ntg) * 32 + lane);
            }
#pragma unroll
            for (int i = 0; i < 4; ++i)
#pragma unroll
                for (int j = 0; j < 4; ++j)
                    acc[i * 4 + j] = __builtin_amdgcn_wmma_f32_16x16x32_f16(
                        false, a[i], false, bb[j], (short)0,
                        acc[i * 4 + j], false, false);
        }

        if (fullTile)
            store_chunk<true >(out, b_out, acc, ch, b, t0, wm, wn, u0, m, half);
        else
            store_chunk<false>(out, b_out, acc, ch, b, t0, wm, wn, u0, m, half);
    }
}

// ---------------------------------------------------------------------------
extern "C" void kernel_launch(void* const* d_in, const int* in_sizes, int n_in,
                              void* d_out, int out_size, void* d_ws, size_t ws_size,
                              hipStream_t stream)
{
    (void)in_sizes; (void)n_in; (void)out_size; (void)ws_size;

    const float* enc    = (const float*)d_in[0];
    const float* pred   = (const float*)d_in[1];
    const float* W_enc  = (const float*)d_in[2];
    const float* b_enc  = (const float*)d_in[3];
    const float* W_pred = (const float*)d_in[4];
    const float* b_pred = (const float*)d_in[5];
    const float* W_out  = (const float*)d_in[6];
    const float* b_out  = (const float*)d_in[7];
    float* out = (float*)d_out;

    char* ws = (char*)d_ws;
    float*    e_ws = (float*)ws;                                   // 1600*512 f32
    float*    p_ws = (float*)(ws + (size_t)RB*RT*RJ*4);            //  800*512 f32
    _Float16* Wf   = (_Float16*)(ws + (size_t)RB*RT*RJ*4
                                    + (size_t)RB*RU*RJ*4);         //  512*1024 f16

    rnnt_proj<<<RB*RT + RB*RU, 256, 0, stream>>>(enc, pred, W_enc, b_enc,
                                                 W_pred, b_pred, e_ws, p_ws);
    rnnt_wswz<<<(16*64*32)/256, 256, 0, stream>>>(W_out, Wf);
    rnnt_joint<<<RB*TG*UT, 256, 0, stream>>>(e_ws, p_ws, Wf, b_out, out);
}
